// BG_ALRT_62921270886438
// MI455X (gfx1250) — matrix operand order repaired and checked by reading, not compiled, not run on hardware
//
#include <hip/hip_runtime.h>
#include <hip/hip_bf16.h>

typedef __bf16 bf16;
typedef __attribute__((ext_vector_type(16))) __bf16 v16bf;
typedef __attribute__((ext_vector_type(8)))  float  v8f;

#define B_ 2
#define T_ 256
#define E_ 512
#define G_ 8
#define GD_ 64
#define L_ 8
#define N_ 64
#define V_ 50257
#define S_ 8
#define BT_ (B_*T_)
#define EPS_ 1.1920928955078125e-07f

// ---- CDNA5 async global->LDS staging (ASYNCcnt path), with safe fallback ----
#if defined(__has_builtin)
#  if __has_builtin(__builtin_amdgcn_global_load_async_to_lds_b128)
#    define HAVE_ASYNC_LDS 1
#  endif
#  if __has_builtin(__builtin_amdgcn_s_wait_asynccnt)
#    define WAIT_ASYNC() __builtin_amdgcn_s_wait_asynccnt(0)
#  endif
#endif
#ifndef HAVE_ASYNC_LDS
#  define HAVE_ASYNC_LDS 0
#endif
#ifndef WAIT_ASYNC
#  define WAIT_ASYNC() asm volatile("s_wait_asynccnt 0" ::: "memory")
#endif

#if HAVE_ASYNC_LDS
// Builtin signature (from compiler diagnostic): param1 = generic int4* (global
// source), param2 = LDS dest, plus two immediate ints (offset, cpol).
typedef int async_v4i __attribute__((vector_size(16)));
#  define STAGE16B(dst, src)                                                   \
     __builtin_amdgcn_global_load_async_to_lds_b128(                           \
        (async_v4i*)(src),                                                     \
        (__attribute__((address_space(3))) async_v4i*)(async_v4i*)(dst), 0, 0)
#  define STAGE_FENCE() do { WAIT_ASYNC(); __syncthreads(); } while (0)
#else
#  define STAGE16B(dst, src) do { *(uint4*)(dst) = *(const uint4*)(src); } while (0)
#  define STAGE_FENCE() __syncthreads()
#endif

// ---------------- WMMA fragment helpers (CDNA5 16x16x32 bf16 layouts) -------
// A (16x32, MxK): element e of the v16bf maps to K = e+16*(lane>>4).
template <typename T>
static __device__ __forceinline__ v16bf load_a_frag(const T* A, int lda, int m0, int k0) {
  int l  = threadIdx.x & 31;
  int m  = m0 + (l & 15);
  int kh = l >> 4;
  const T* row = A + (size_t)m * lda + k0 + kh * 16;
  v16bf a;
#pragma unroll
  for (int e = 0; e < 16; ++e) a[e] = (bf16)row[e];
  return a;
}

// B (32x16, KxN): column n of B = contiguous row n of row-major W[N][K].
template <typename T>
static __device__ __forceinline__ v16bf load_b_rows(const T* W, int ldw, int n0, int k0) {
  int l  = threadIdx.x & 31;
  int n  = n0 + (l & 15);
  int kh = l >> 4;
  const T* row = W + (size_t)n * ldw + k0 + kh * 16;
  v16bf b;
#pragma unroll
  for (int e = 0; e < 16; ++e) b[e] = (bf16)row[e];
  return b;
}

// B fragment with strided K:  B[k][n] = src[k*strideK + n]
template <typename T>
static __device__ __forceinline__ v16bf load_b_strided(const T* src, size_t strideK, int n0, int k0) {
  int l  = threadIdx.x & 31;
  int n  = n0 + (l & 15);
  int kh = l >> 4;
  const T* p = src + (size_t)(k0 + kh * 16) * strideK + n;
  v16bf b;
#pragma unroll
  for (int e = 0; e < 16; ++e) b[e] = (bf16)p[(size_t)e * strideK];
  return b;
}

static __device__ __forceinline__ v8f wmma_bf16(v16bf a, v16bf b, v8f c) {
  return __builtin_amdgcn_wmma_f32_16x16x32_bf16(false, a, false, b, (short)0, c, false, false);
}

// ---------------- setup: depths iteration, rope tables, degenerate-einsum rowsums
__global__ void k_setup(const float* dep, const float* attn_proj, const float* mlp_proj,
                        float* depths, float* costab, float* sintab,
                        float* rs_attn, float* rs_mlp) {
  __shared__ float d[N_];
  int tid = threadIdx.x;
  if (tid < N_) d[tid] = 0.f;
  __syncthreads();
  for (int it = 0; it < L_; ++it) {
    float s = 0.f;
    if (tid < N_) {
      for (int m = 0; m < N_; ++m) {
        float w = dep[tid * N_ + m];
        w = w > 0.f ? w : 0.f;
        s += w * (d[m] + 1.f);
      }
    }
    __syncthreads();
    if (tid < N_) d[tid] = s;
    __syncthreads();
  }
  if (tid < N_) depths[tid] = d[tid];
  for (int i = tid; i < T_ * 32; i += blockDim.x) {
    int t = i / 32, f = i % 32;
    double invf = pow(10000.0, -((double)(2 * f)) / (double)GD_);
    double ang  = (double)t * invf;
    costab[i] = (float)cos(ang);
    sintab[i] = (float)sin(ang);
  }
  for (int i = tid; i < N_ * GD_; i += blockDim.x) {
    int n = i / GD_, g = i % GD_;
    float sa = 0.f;
    const float* ap = attn_proj + ((size_t)n * GD_ + g) * GD_;
    for (int k = 0; k < GD_; ++k) sa += ap[k];
    rs_attn[i] = sa;
    float sm = 0.f;
    const float* mp = mlp_proj + ((size_t)n * GD_ + g) * (4 * GD_);
    for (int k = 0; k < 4 * GD_; ++k) sm += mp[k];
    rs_mlp[i] = sm;
  }
}

__global__ void k_cvt(const float* src, bf16* dst, int nelem) {
  int i = blockIdx.x * blockDim.x + threadIdx.x;
  if (i < nelem) dst[i] = (bf16)src[i];
}

// x = rms(wte[idx]); p_cont = 1
__global__ void k_embed(const int* idx, const float* wte, float* x, float* p_cont) {
  int bt = blockIdx.x, tid = threadIdx.x;
  const float* row = wte + (size_t)idx[bt] * E_;
  float v0 = row[tid], v1 = row[tid + 256];
  __shared__ float red[256];
  red[tid] = v0 * v0 + v1 * v1;
  __syncthreads();
  for (int s = 128; s > 0; s >>= 1) { if (tid < s) red[tid] += red[tid + s]; __syncthreads(); }
  float r = rsqrtf(red[0] / E_ + EPS_);
  x[(size_t)bt * E_ + tid]       = v0 * r;
  x[(size_t)bt * E_ + tid + 256] = v1 * r;
  if (tid == 0) p_cont[bt] = 1.f;
}

__global__ void k_stepprep(const float* x, bf16* xb, float* ssum) {
  int i = blockIdx.x * blockDim.x + threadIdx.x;
  if (i < BT_ * E_) xb[i] = (bf16)x[i];
  if (i < BT_ * N_) ssum[i] = 0.f;
}

// xi[bt,n,g] = sum_e x[bt,e]*adapters[n,g,e] + x[bt,(n%G)*GD+g]
// Block: one adapter n shared by 8 waves; B tile staged in LDS via async copy.
__global__ void k_xi(const bf16* xb, const bf16* adp, const float* x, float* xi, bf16* xib) {
  __shared__ bf16 wtile[2][GD_ * 32];          // 2 x 4KB double buffer
  int n  = blockIdx.x >> 2;
  int mt = (blockIdx.x & 3) * 8 + (threadIdx.x >> 5);
  int m0 = mt * 16;
  int tid = threadIdx.x;
  int row = tid >> 2, chunk = tid & 3;         // 64 rows x 4 x 16B = 4KB tile
  const bf16* gsrc = adp + (size_t)n * GD_ * E_ + (size_t)row * E_ + chunk * 8;

  STAGE16B(&wtile[0][row * 32 + chunk * 8], gsrc);
  STAGE_FENCE();

  v8f acc[4] = {};
  for (int kk = 0; kk < 16; ++kk) {
    int k0 = kk * 32;
    if (kk + 1 < 16)
      STAGE16B(&wtile[(kk + 1) & 1][row * 32 + chunk * 8], gsrc + k0 + 32);
    const bf16* Bt = &wtile[kk & 1][0];
    v16bf a  = load_a_frag(xb, E_, m0, k0);
    v16bf b0 = load_b_rows(Bt, 32, 0,  0);
    v16bf b1 = load_b_rows(Bt, 32, 16, 0);
    v16bf b2 = load_b_rows(Bt, 32, 32, 0);
    v16bf b3 = load_b_rows(Bt, 32, 48, 0);
    acc[0] = wmma_bf16(a, b0, acc[0]);
    acc[1] = wmma_bf16(a, b1, acc[1]);
    acc[2] = wmma_bf16(a, b2, acc[2]);
    acc[3] = wmma_bf16(a, b3, acc[3]);
    STAGE_FENCE();
  }
  int l = threadIdx.x & 31, half = l >> 4, col = l & 15, grp = n & (G_ - 1);
#pragma unroll
  for (int j = 0; j < 4; ++j) {
    int g = j * 16 + col;
#pragma unroll
    for (int r = 0; r < 8; ++r) {
      int m = m0 + r + half * 8;
      float v = acc[j][r] + x[(size_t)m * E_ + grp * GD_ + g];
      size_t o = (size_t)m * (N_ * GD_) + n * GD_ + g;
      xi[o] = v;
      xib[o] = (bf16)v;
    }
  }
}

// qkv[bt,n,o] = sum_g xi[bt,n,g]*qkv_w[n,o,g]   (WMMA, o in 0..191)
__global__ void k_qkv(const bf16* xib, const bf16* qw, float* qkv) {
  int wid = (blockIdx.x * blockDim.x + threadIdx.x) >> 5;   // 6144 waves
  int n = wid / 96, rem = wid % 96;
  int mt = rem / 3, oq = rem % 3;
  int m0 = mt * 16;
  const bf16* A = xib + n * GD_;
  const bf16* W = qw + (size_t)n * 3 * GD_ * GD_;
  v8f acc[4] = {};
  for (int k0 = 0; k0 < GD_; k0 += 32) {
    v16bf a  = load_a_frag(A, N_ * GD_, m0, k0);
    v16bf b0 = load_b_rows(W, GD_, oq * 64 + 0,  k0);
    v16bf b1 = load_b_rows(W, GD_, oq * 64 + 16, k0);
    v16bf b2 = load_b_rows(W, GD_, oq * 64 + 32, k0);
    v16bf b3 = load_b_rows(W, GD_, oq * 64 + 48, k0);
    acc[0] = wmma_bf16(a, b0, acc[0]);
    acc[1] = wmma_bf16(a, b1, acc[1]);
    acc[2] = wmma_bf16(a, b2, acc[2]);
    acc[3] = wmma_bf16(a, b3, acc[3]);
  }
  int l = threadIdx.x & 31, half = l >> 4, col = l & 15;
#pragma unroll
  for (int j = 0; j < 4; ++j) {
    int o = oq * 64 + j * 16 + col;
#pragma unroll
    for (int r = 0; r < 8; ++r) {
      int m = m0 + r + half * 8;
      qkv[(size_t)m * (N_ * 3 * GD_) + n * (3 * GD_) + o] = acc[j][r];
    }
  }
}

// rope + rms on q,k; cvt v -> bf16. one wave per (bt,n)
__global__ void k_rope(const float* qkv, const float* costab, const float* sintab,
                       bf16* qb, bf16* kb, bf16* vb) {
  int wid = (blockIdx.x * blockDim.x + threadIdx.x) >> 5;
  int l = threadIdx.x & 31;
  int bt = wid >> 6, n = wid & 63;
  int tpos = bt & (T_ - 1);
  const float* base = qkv + (size_t)bt * (N_ * 3 * GD_) + n * (3 * GD_);
  float c = costab[tpos * 32 + l], s = sintab[tpos * 32 + l];
  size_t ob = (size_t)bt * (N_ * GD_) + n * GD_;
  {
    float x1 = base[l], x2 = base[l + 32];
    float r1 = x1 * c + x2 * s, r2 = -x1 * s + x2 * c;
    float ss = r1 * r1 + r2 * r2;
    for (int m = 16; m >= 1; m >>= 1) ss += __shfl_xor(ss, m, 32);
    float r = rsqrtf(ss / GD_ + EPS_);
    qb[ob + l] = (bf16)(r1 * r);
    qb[ob + l + 32] = (bf16)(r2 * r);
  }
  {
    float x1 = base[GD_ + l], x2 = base[GD_ + l + 32];
    float r1 = x1 * c + x2 * s, r2 = -x1 * s + x2 * c;
    float ss = r1 * r1 + r2 * r2;
    for (int m = 16; m >= 1; m >>= 1) ss += __shfl_xor(ss, m, 32);
    float r = rsqrtf(ss / GD_ + EPS_);
    kb[ob + l] = (bf16)(r1 * r);
    kb[ob + l + 32] = (bf16)(r2 * r);
  }
  vb[ob + l]      = (bf16)base[2 * GD_ + l];
  vb[ob + l + 32] = (bf16)base[2 * GD_ + l + 32];
}

// scores[b,n,t,s] = scale * q.k  with causal mask  (WMMA)
__global__ void k_scores(const bf16* qb, const bf16* kb, float* scores) {
  int wid = (blockIdx.x * blockDim.x + threadIdx.x) >> 5;
  int bn = wid >> 8, tp = wid & 255;
  int tt = tp >> 4, st = tp & 15;
  int b = bn >> 6, n = bn & 63;
  int t0 = tt * 16, s0 = st * 16;
  const bf16* Q = qb + (size_t)b * T_ * (N_ * GD_) + n * GD_;
  const bf16* K = kb + (size_t)b * T_ * (N_ * GD_) + n * GD_;
  v8f acc = {};
  if (st <= tt) {   // uniform per wave
    v16bf a0 = load_a_frag(Q, N_ * GD_, t0, 0);
    v16bf c0 = load_b_rows(K, N_ * GD_, s0, 0);
    v16bf a1 = load_a_frag(Q, N_ * GD_, t0, 32);
    v16bf c1 = load_b_rows(K, N_ * GD_, s0, 32);
    acc = wmma_bf16(a0, c0, acc);
    acc = wmma_bf16(a1, c1, acc);
  }
  int l = threadIdx.x & 31, half = l >> 4, col = l & 15;
  float* out = scores + (size_t)bn * T_ * T_;
#pragma unroll
  for (int r = 0; r < 8; ++r) {
    int t = t0 + r + half * 8, s = s0 + col;
    out[(size_t)t * T_ + s] = (s <= t) ? acc[r] * 0.125f : -1e30f;
  }
}

__global__ void k_softmax(const float* scores, bf16* P) {
  int wid = (blockIdx.x * blockDim.x + threadIdx.x) >> 5;
  int l = threadIdx.x & 31;
  const float* row = scores + (size_t)wid * T_;
  float v[8], mx = -1e30f;
#pragma unroll
  for (int i = 0; i < 8; ++i) { v[i] = row[l + i * 32]; mx = fmaxf(mx, v[i]); }
  for (int m = 16; m >= 1; m >>= 1) mx = fmaxf(mx, __shfl_xor(mx, m, 32));
  float sum = 0.f;
#pragma unroll
  for (int i = 0; i < 8; ++i) { v[i] = __expf(v[i] - mx); sum += v[i]; }
  for (int m = 16; m >= 1; m >>= 1) sum += __shfl_xor(sum, m, 32);
  float inv = 1.f / sum;
  bf16* orow = P + (size_t)wid * T_;
#pragma unroll
  for (int i = 0; i < 8; ++i) orow[l + i * 32] = (bf16)(v[i] * inv);
}

// att = P @ v ; fused: xi_mid = xi + att * rowsum(attn_proj)   (WMMA)
__global__ void k_att(const bf16* P, const bf16* vb, const float* xi,
                      const float* rs_attn, float* xi_mid) {
  int wid = (blockIdx.x * blockDim.x + threadIdx.x) >> 5;
  int bn = wid >> 4, tt = wid & 15;
  int b = bn >> 6, n = bn & 63;
  int t0 = tt * 16;
  const bf16* A = P + (size_t)bn * T_ * T_;
  const bf16* Vs = vb + (size_t)b * T_ * (N_ * GD_) + n * GD_;
  v8f acc[4] = {};
  for (int k0 = 0; k0 < T_; k0 += 32) {
    v16bf a  = load_a_frag(A, T_, t0, k0);
    v16bf b0 = load_b_strided(Vs, (size_t)(N_ * GD_), 0,  k0);
    v16bf b1 = load_b_strided(Vs, (size_t)(N_ * GD_), 16, k0);
    v16bf b2 = load_b_strided(Vs, (size_t)(N_ * GD_), 32, k0);
    v16bf b3 = load_b_strided(Vs, (size_t)(N_ * GD_), 48, k0);
    acc[0] = wmma_bf16(a, b0, acc[0]);
    acc[1] = wmma_bf16(a, b1, acc[1]);
    acc[2] = wmma_bf16(a, b2, acc[2]);
    acc[3] = wmma_bf16(a, b3, acc[3]);
  }
  int l = threadIdx.x & 31, half = l >> 4, col = l & 15;
#pragma unroll
  for (int j = 0; j < 4; ++j) {
    int g = j * 16 + col;
    float rs = rs_attn[n * GD_ + g];
#pragma unroll
    for (int r = 0; r < 8; ++r) {
      int t = t0 + r + half * 8;
      size_t o = ((size_t)b * T_ + t) * (N_ * GD_) + n * GD_ + g;
      xi_mid[o] = xi[o] + acc[j][r] * rs;
    }
  }
}

__global__ void k_rmsmid(const float* xi_mid, bf16* hb) {
  int wid = (blockIdx.x * blockDim.x + threadIdx.x) >> 5;
  int l = threadIdx.x & 31;
  const float* row = xi_mid + (size_t)wid * GD_;
  float a = row[l], b = row[l + 32];
  float ss = a * a + b * b;
  for (int m = 16; m >= 1; m >>= 1) ss += __shfl_xor(ss, m, 32);
  float r = rsqrtf(ss / GD_ + EPS_);
  bf16* o = hb + (size_t)wid * GD_;
  o[l] = (bf16)(a * r);
  o[l + 32] = (bf16)(b * r);
}

// fc = rms(xi_mid) @ mlp_fc^T ; reduce sum_o relu(fc)^2 into ssum[bt,n]
__global__ void k_fc(const bf16* hb, const bf16* fcw, float* ssum) {
  int wid = (blockIdx.x * blockDim.x + threadIdx.x) >> 5;
  int n = wid >> 7, rem = wid & 127;
  int mt = rem >> 2, oq = rem & 3;
  int m0 = mt * 16;
  const bf16* A = hb + n * GD_;
  const bf16* W = fcw + (size_t)n * 4 * GD_ * GD_;
  v8f acc[4] = {};
  for (int k0 = 0; k0 < GD_; k0 += 32) {
    v16bf a  = load_a_frag(A, N_ * GD_, m0, k0);
    v16bf b0 = load_b_rows(W, GD_, oq * 64 + 0,  k0);
    v16bf b1 = load_b_rows(W, GD_, oq * 64 + 16, k0);
    v16bf b2 = load_b_rows(W, GD_, oq * 64 + 32, k0);
    v16bf b3 = load_b_rows(W, GD_, oq * 64 + 48, k0);
    acc[0] = wmma_bf16(a, b0, acc[0]);
    acc[1] = wmma_bf16(a, b1, acc[1]);
    acc[2] = wmma_bf16(a, b2, acc[2]);
    acc[3] = wmma_bf16(a, b3, acc[3]);
  }
  int l = threadIdx.x & 31, half = l >> 4;
#pragma unroll
  for (int r = 0; r < 8; ++r) {
    float s = 0.f;
#pragma unroll
    for (int j = 0; j < 4; ++j) { float v = fmaxf(acc[j][r], 0.f); s += v * v; }
    for (int m = 8; m >= 1; m >>= 1) s += __shfl_xor(s, m, 32);
    if ((l & 15) == 0) {
      int mrow = m0 + r + half * 8;
      atomicAdd(&ssum[(size_t)mrow * N_ + n], s);
    }
  }
}

// x += p_cont * sum_l w_eff(n) * (xi_mid + ssum*rowsum_mlp - xi)
__global__ void k_update(const float* xi, const float* xi_mid, const float* ssum,
                         const float* rs_mlp, const float* depths, const float* p_cont,
                         float td, float* x) {
  int i = blockIdx.x * blockDim.x + threadIdx.x;
  int bt = i >> 9, e = i & 511;
  int grp = e >> 6, g = e & 63;
  float acc = 0.f;
#pragma unroll
  for (int ly = 0; ly < L_; ++ly) {
    int n = ly * G_ + grp;
    float w = __expf(-fabsf(depths[n] - td));
    w = (w > 0.15f) ? w : 0.f;
    size_t o = (size_t)bt * (N_ * GD_) + n * GD_ + g;
    float up = xi_mid[o] + ssum[(size_t)bt * N_ + n] * rs_mlp[n * GD_ + g] - xi[o];
    acc += up * w;
  }
  x[i] += acc * p_cont[bt];
}

__global__ void k_router(const float* x, const float* rw, const float* rb, float* p_cont) {
  int wid = (blockIdx.x * blockDim.x + threadIdx.x) >> 5;
  int l = threadIdx.x & 31;
  const float* row = x + (size_t)wid * E_;
  float s = 0.f;
#pragma unroll
  for (int i = 0; i < 16; ++i) s += row[l + i * 32] * rw[l + i * 32];
  for (int m = 16; m >= 1; m >>= 1) s += __shfl_xor(s, m, 32);
  if (l == 0) {
    float ph = 1.f / (1.f + __expf(-(s + rb[0])));
    p_cont[wid] *= (1.f - ph);
  }
}

__global__ void k_finalrms(const float* x, bf16* xfb) {
  int bt = blockIdx.x, tid = threadIdx.x;
  const float* row = x + (size_t)bt * E_;
  float v0 = row[tid], v1 = row[tid + 256];
  __shared__ float red[256];
  red[tid] = v0 * v0 + v1 * v1;
  __syncthreads();
  for (int s = 128; s > 0; s >>= 1) { if (tid < s) red[tid] += red[tid + s]; __syncthreads(); }
  float r = rsqrtf(red[0] / E_ + EPS_);
  xfb[(size_t)bt * E_ + tid]       = (bf16)(v0 * r);
  xfb[(size_t)bt * E_ + tid + 256] = (bf16)(v1 * r);
}

// logits = rms(x) @ lm_head^T ; out = 15*tanh(logits/15)
// Block: one 64-wide vocab tile shared by 8 waves; f32 weights converted to
// bf16 while staging into a double-buffered LDS tile.
__global__ void k_lmhead(const bf16* xfb, const float* lm, float* out) {
  __shared__ bf16 ltile[2][GD_ * 32];          // 2 x 4KB
  int q  = blockIdx.x >> 2;                    // 0..785 vocab quad
  int mt = (blockIdx.x & 3) * 8 + (threadIdx.x >> 5);
  int m0 = mt * 16;
  int v0q = q * 64;
  int tid = threadIdx.x;
  int row = tid >> 2, chunk = tid & 3;
  int vr = v0q + row; if (vr >= V_) vr = V_ - 1;
  const float* gsrc = lm + (size_t)vr * E_ + chunk * 8;
  {
    bf16* d = &ltile[0][row * 32 + chunk * 8];
#pragma unroll
    for (int e = 0; e < 8; ++e) d[e] = (bf16)gsrc[e];
  }
  __syncthreads();
  v8f acc[4] = {};
  for (int kk = 0; kk < 16; ++kk) {
    int k0 = kk * 32;
    if (kk + 1 < 16) {
      const float* sp = gsrc + k0 + 32;
      bf16* d = &ltile[(kk + 1) & 1][row * 32 + chunk * 8];
#pragma unroll
      for (int e = 0; e < 8; ++e) d[e] = (bf16)sp[e];
    }
    const bf16* Bt = &ltile[kk & 1][0];
    v16bf a  = load_a_frag(xfb, E_, m0, k0);
    v16bf b0 = load_b_rows(Bt, 32, 0,  0);
    v16bf b1 = load_b_rows(Bt, 32, 16, 0);
    v16bf b2 = load_b_rows(Bt, 32, 32, 0);
    v16bf b3 = load_b_rows(Bt, 32, 48, 0);
    acc[0] = wmma_bf16(a, b0, acc[0]);
    acc[1] = wmma_bf16(a, b1, acc[1]);
    acc[2] = wmma_bf16(a, b2, acc[2]);
    acc[3] = wmma_bf16(a, b3, acc[3]);
    __syncthreads();
  }
  int l = threadIdx.x & 31, half = l >> 4, col = l & 15;
#pragma unroll
  for (int j = 0; j < 4; ++j) {
    int v = v0q + j * 16 + col;
    if (v < V_) {
#pragma unroll
      for (int r = 0; r < 8; ++r) {
        int m = m0 + r + half * 8;
        out[(size_t)m * V_ + v] = 15.f * tanhf(acc[j][r] * (1.f / 15.f));
      }
    }
  }
}

extern "C" void kernel_launch(void* const* d_in, const int* in_sizes, int n_in,
                              void* d_out, int out_size, void* d_ws, size_t ws_size,
                              hipStream_t stream) {
  (void)in_sizes; (void)n_in; (void)out_size; (void)ws_size;
  const int*   idx       = (const int*)  d_in[0];
  // d_in[1] = n_steps device scalar; host trip count must be static -> S_ (== 8)
  const float* wte       = (const float*)d_in[2];
  const float* adapters  = (const float*)d_in[3];
  const float* qkv_w     = (const float*)d_in[4];
  const float* attn_proj = (const float*)d_in[5];
  const float* mlp_fc    = (const float*)d_in[6];
  const float* mlp_proj  = (const float*)d_in[7];
  const float* dep       = (const float*)d_in[8];
  const float* router_w  = (const float*)d_in[9];
  const float* router_b  = (const float*)d_in[10];
  const float* lm_head   = (const float*)d_in[11];
  float* out = (float*)d_out;

  char* p = (char*)d_ws;
  auto alloc = [&](size_t bytes) { char* r = p; p += (bytes + 255) & ~(size_t)255; return r; };
  float* x      = (float*)alloc((size_t)BT_ * E_ * 4);
  bf16*  xb     = (bf16*) alloc((size_t)BT_ * E_ * 2);
  float* depths = (float*)alloc(N_ * 4);
  float* costab = (float*)alloc((size_t)T_ * 32 * 4);
  float* sintab = (float*)alloc((size_t)T_ * 32 * 4);
  float* rs_attn= (float*)alloc((size_t)N_ * GD_ * 4);
  float* rs_mlp = (float*)alloc((size_t)N_ * GD_ * 4);
  float* p_cont = (float*)alloc(BT_ * 4);
  bf16*  adp_b  = (bf16*) alloc((size_t)N_ * GD_ * E_ * 2);
  bf16*  qkvw_b = (bf16*) alloc((size_t)N_ * 3 * GD_ * GD_ * 2);
  bf16*  fcw_b  = (bf16*) alloc((size_t)N_ * 4 * GD_ * GD_ * 2);
  float* xi     = (float*)alloc((size_t)BT_ * N_ * GD_ * 4);
  bf16*  xib    = (bf16*) alloc((size_t)BT_ * N_ * GD_ * 2);
  float* qkv    = (float*)alloc((size_t)BT_ * N_ * 3 * GD_ * 4);
  bf16*  qb     = (bf16*) alloc((size_t)BT_ * N_ * GD_ * 2);
  bf16*  kb     = (bf16*) alloc((size_t)BT_ * N_ * GD_ * 2);
  bf16*  vb     = (bf16*) alloc((size_t)BT_ * N_ * GD_ * 2);
  float* scores = (float*)alloc((size_t)B_ * N_ * T_ * T_ * 4);
  bf16*  Pm     = (bf16*) alloc((size_t)B_ * N_ * T_ * T_ * 2);
  float* xi_mid = (float*)alloc((size_t)BT_ * N_ * GD_ * 4);
  bf16*  hb     = (bf16*) alloc((size_t)BT_ * N_ * GD_ * 2);
  float* ssum   = (float*)alloc((size_t)BT_ * N_ * 4);
  bf16*  xfb    = (bf16*) alloc((size_t)BT_ * E_ * 2);

  k_setup<<<1, 256, 0, stream>>>(dep, attn_proj, mlp_proj, depths, costab, sintab, rs_attn, rs_mlp);
  k_cvt<<<(N_*GD_*E_ + 255)/256, 256, 0, stream>>>(adapters, adp_b, N_*GD_*E_);
  k_cvt<<<(N_*3*GD_*GD_ + 255)/256, 256, 0, stream>>>(qkv_w, qkvw_b, N_*3*GD_*GD_);
  k_cvt<<<(N_*4*GD_*GD_ + 255)/256, 256, 0, stream>>>(mlp_fc, fcw_b, N_*4*GD_*GD_);
  k_embed<<<BT_, 256, 0, stream>>>(idx, wte, x, p_cont);

  for (int t = 0; t < S_; ++t) {
    float td = (float)t * ((float)L_ / (float)S_);
    k_stepprep<<<(BT_*E_)/256, 256, 0, stream>>>(x, xb, ssum);
    k_xi     <<<256,  256, 0, stream>>>(xb, adp_b, x, xi, xib);
    k_qkv    <<<768,  256, 0, stream>>>(xib, qkvw_b, qkv);
    k_rope   <<<4096, 256, 0, stream>>>(qkv, costab, sintab, qb, kb, vb);
    k_scores <<<4096, 256, 0, stream>>>(qb, kb, scores);
    k_softmax<<<4096, 256, 0, stream>>>(scores, Pm);
    k_att    <<<256,  256, 0, stream>>>(Pm, vb, xi, rs_attn, xi_mid);
    k_rmsmid <<<4096, 256, 0, stream>>>(xi_mid, hb);
    k_fc     <<<1024, 256, 0, stream>>>(hb, fcw_b, ssum);
    k_update <<<(BT_*E_)/256, 256, 0, stream>>>(xi, xi_mid, ssum, rs_mlp, depths, p_cont, td, x);
    k_router <<<64,   256, 0, stream>>>(x, router_w, router_b, p_cont);
  }

  k_finalrms<<<BT_, 256, 0, stream>>>(x, xfb);
  k_lmhead  <<<3144, 256, 0, stream>>>(xfb, lm_head, out);
}
